// NNLREPL_78271484003075
// MI455X (gfx1250) — matrix operand order repaired and checked by reading, not compiled, not run on hardware
//
#include <hip/hip_runtime.h>
#include <hip/hip_bf16.h>

// MI455X / gfx1250, wave32. bf16 WMMA (16x16x32, f32 accum) flash-attention.
// All fragment loads are 16-byte vector loads from K-contiguous layouts.

typedef __attribute__((ext_vector_type(16))) __bf16 v16bf;
typedef __attribute__((ext_vector_type(8)))  __bf16 v8bf;
typedef __attribute__((ext_vector_type(8)))  float  v8f;

static __device__ __forceinline__ __bf16 f2bf(float f) { return (__bf16)f; }

static __device__ __forceinline__ v16bf concat8(v8bf lo, v8bf hi) {
    return __builtin_shufflevector(lo, hi, 0,1,2,3,4,5,6,7,8,9,10,11,12,13,14,15);
}

// A-frag (16x32 MxK, 16-bit), row-major source: lane L -> row M=L&15,
// K runs [kb,kb+8) and [kb+16,kb+24), kb = k0 + 8*(L>>4).  Two b128 loads.
static __device__ __forceinline__ v16bf load_afrag(const __bf16* p, int rowStride,
                                                   int k0, int lane) {
    const __bf16* q = p + (lane & 15) * rowStride + k0 + ((lane >> 4) << 3);
    return concat8(*(const v8bf*)q, *(const v8bf*)(q + 16));
}
// B-frag (32x16 KxN) from an N-major ("[N][K]") source: lane L -> col N=col0+(L&15),
// K runs [kb,kb+8) and [kb+16,kb+24), kb = k0 + 8*(L>>4).  Two b128 loads.
static __device__ __forceinline__ v16bf load_bfragT(const __bf16* p, int rowStride,
                                                    int col0, int k0, int lane) {
    const __bf16* q = p + (col0 + (lane & 15)) * rowStride + k0 + ((lane >> 4) << 3);
    return concat8(*(const v8bf*)q, *(const v8bf*)(q + 16));
}

// ---- Kernel 0: fuse weights (bf16) + biases (f32): rows [0,64)=wq, [64,128)=wk,
// [128,384)=wv.  One block per fused row.
__global__ void k_prep(const float* __restrict__ wq, const float* __restrict__ bq,
                       const float* __restrict__ wk, const float* __restrict__ bk,
                       const float* __restrict__ wv, const float* __restrict__ bv,
                       __bf16* __restrict__ wf, float* __restrict__ bfused) {
    const int m = blockIdx.x, c = threadIdx.x;
    const float* wrow = (m < 64) ? (wq + m * 256)
                      : (m < 128) ? (wk + (m - 64) * 256)
                                  : (wv + (m - 128) * 256);
    wf[m * 256 + c] = f2bf(wrow[c]);
    if (c == 0)
        bfused[m] = (m < 64) ? bq[m] : (m < 128) ? bk[m - 64] : bv[m - 128];
}

// ---- Kernel 1: x[b][c][n] (f32) -> xb[b][n][c] (bf16), 64x64 LDS transpose ----
__global__ void k_cvt_t(const float* __restrict__ x, __bf16* __restrict__ xb) {
    __shared__ __bf16 t[64][66];                 // 33-dword rows, conflict-free
    const int tid = threadIdx.x;
    const int b = blockIdx.z, c0 = blockIdx.y << 6, n0 = blockIdx.x << 6;
    #pragma unroll
    for (int p = 0; p < 16; ++p) {               // read coalesced along n
        int i = p * 256 + tid, r = i >> 6, c = i & 63;
        t[r][c] = f2bf(x[((size_t)(b << 8) + c0 + r) * 4096 + n0 + c]);
    }
    __syncthreads();
    #pragma unroll
    for (int p = 0; p < 16; ++p) {               // write coalesced along c
        int i = p * 256 + tid, rn = i >> 6, cc = i & 63;
        xb[((size_t)(b << 12) + n0 + rn) * 256 + c0 + cc] = t[cc][rn];
    }
}

// ---- Kernel 2: fused QKV projection -----------------------------------------
// Per wave: 16(M)x64(N) tile, K-loop 256 in steps of 32.  Outputs (bf16):
// q_ws[b][n][64], k_ws[b][n][64] (both n-major), v_ws[b][c][n].
__global__ void __launch_bounds__(256)
k_qkv(const __bf16* __restrict__ xb, const __bf16* __restrict__ wf,
      const float* __restrict__ bfused,
      __bf16* __restrict__ qws, __bf16* __restrict__ kws, __bf16* __restrict__ vws) {
    const int tid  = threadIdx.x;
    const int lane = tid & 31, wave = tid >> 5;
    const int b  = blockIdx.z;
    const int m0 = blockIdx.y << 4;
    const int n0 = (blockIdx.x << 9) + (wave << 6);
    const int cc = lane & 15, hh = lane >> 4;

    const __bf16* abase = wf + m0 * 256;                       // A: fused W rows
    const __bf16* bbase = xb + ((size_t)b << 12) * 256;        // B: x, n-major

    v8f acc[4] = {};
    #pragma unroll
    for (int k0 = 0; k0 < 256; k0 += 32) {
        v16bf a = load_afrag(abase, 256, k0, lane);
        #pragma unroll
        for (int j = 0; j < 4; ++j) {
            v16bf bx = load_bfragT(bbase, 256, n0 + (j << 4), k0, lane);
            acc[j] = __builtin_amdgcn_wmma_f32_16x16x32_bf16(
                false, a, false, bx, (short)0, acc[j], false, false);
        }
    }
    if (m0 < 64) {
        #pragma unroll
        for (int j = 0; j < 4; ++j)
            #pragma unroll
            for (int r = 0; r < 8; ++r) {
                int m = m0 + r + (hh << 3), n = n0 + (j << 4) + cc;
                qws[((size_t)(b << 12) + n) * 64 + m] = f2bf(acc[j][r] + bfused[m]);
            }
    } else if (m0 < 128) {
        #pragma unroll
        for (int j = 0; j < 4; ++j)
            #pragma unroll
            for (int r = 0; r < 8; ++r) {
                int m = m0 + r + (hh << 3), n = n0 + (j << 4) + cc;
                kws[((size_t)(b << 12) + n) * 64 + (m - 64)] = f2bf(acc[j][r] + bfused[m]);
            }
    } else {
        #pragma unroll
        for (int j = 0; j < 4; ++j)
            #pragma unroll
            for (int r = 0; r < 8; ++r) {
                int m = m0 + r + (hh << 3), n = n0 + (j << 4) + cc;
                vws[((size_t)(b << 8) + (m - 128)) * 4096 + n] = f2bf(acc[j][r] + bfused[m]);
            }
    }
}

// ---- Kernel 3: flash attention + residual -----------------------------------
// Block = 8 waves; wave owns 16 query rows.  64 keys/step (softmax + O-rescale
// amortized over 40 wmma):
//   S = Q(16x64) x K(64x64)  (8 wmma), online softmax in registers,
//   P via per-wave LDS transpose (same-wave DS in-order: s_wait_dscnt 0),
//   O(16x256) += P(16x64) x V^T(64x256)  (32 wmma).
__global__ void __launch_bounds__(256)
k_attn(const float* __restrict__ x,
       const __bf16* __restrict__ qws, const __bf16* __restrict__ kws,
       const __bf16* __restrict__ vws, const float* __restrict__ gamma,
       float* __restrict__ out) {
    __shared__ alignas(32) __bf16 ldsK[64 * 72];    // [m][cq], 36-dword rows
    __shared__ alignas(32) __bf16 ldsV[256 * 72];   // [c][m],  36-dword rows
    __shared__ alignas(32) __bf16 ldsP[8 * 16 * 64];

    const int tid  = threadIdx.x;
    const int lane = tid & 31, wave = tid >> 5;
    const int b   = blockIdx.y;
    const int nq0 = (blockIdx.x << 7) + (wave << 4);
    const int cc  = lane & 15, hh = lane >> 4;

    const __bf16* qbase = qws + ((size_t)(b << 12) + nq0) * 64;
    v16bf aQ[2];
    aQ[0] = load_afrag(qbase, 64, 0,  lane);
    aQ[1] = load_afrag(qbase, 64, 32, lane);

    float mi[8], li[8];
    #pragma unroll
    for (int r = 0; r < 8; ++r) { mi[r] = -1e30f; li[r] = 0.0f; }
    v8f acc[16] = {};

    for (int m0 = 0; m0 < 4096; m0 += 64) {
        __syncthreads();
        // Stage K tile [64 m][64 cq] from n-major kws: 2x b128 per thread.
        #pragma unroll
        for (int i = 0; i < 2; ++i) {
            int idx = tid + (i << 8);
            int mm = idx >> 3, k = (idx & 7) << 3;
            *(v8bf*)&ldsK[mm * 72 + k] =
                *(const v8bf*)&kws[((size_t)(b << 12) + m0 + mm) * 64 + k];
        }
        // Stage V tile [256 c][64 m] from vws (m-contiguous rows): 8x b128.
        #pragma unroll
        for (int i = 0; i < 8; ++i) {
            int idx = tid + (i << 8);
            int c = idx >> 3, q8 = (idx & 7) << 3;
            *(v8bf*)&ldsV[c * 72 + q8] =
                *(const v8bf*)&vws[((size_t)(b << 8) + c) * 4096 + m0 + q8];
        }
        // Prefetch next step's tiles into cache while this step computes.
        if (m0 + 64 < 4096) {
            __builtin_prefetch(&vws[((size_t)(b << 8) + tid) * 4096 + m0 + 64], 0, 1);
            if (tid < 64)
                __builtin_prefetch(&kws[((size_t)(b << 12) + m0 + 64 + tid) * 64], 0, 1);
        }
        __syncthreads();

        // Energy: four 16x16 D-tiles over keys [m0, m0+64).
        v8f s[4] = {};
        #pragma unroll
        for (int kt = 0; kt < 2; ++kt)
            #pragma unroll
            for (int j = 0; j < 4; ++j) {
                v16bf bk16 = load_bfragT(ldsK, 72, j << 4, kt << 5, lane);
                s[j] = __builtin_amdgcn_wmma_f32_16x16x32_bf16(
                    false, aQ[kt], false, bk16, (short)0, s[j], false, false);
            }

        // Online softmax: row r of a D-tile lives in VGPR r across a 16-lane half.
        float p[4][8], scale[8];
        #pragma unroll
        for (int r = 0; r < 8; ++r) {
            float mv = fmaxf(fmaxf(s[0][r], s[1][r]), fmaxf(s[2][r], s[3][r]));
            #pragma unroll
            for (int off = 1; off < 16; off <<= 1)
                mv = fmaxf(mv, __shfl_xor(mv, off, 32));
            const float mnew = fmaxf(mi[r], mv);
            float rs = 0.0f;
            #pragma unroll
            for (int j = 0; j < 4; ++j) {
                p[j][r] = __expf(s[j][r] - mnew);
                rs += p[j][r];
            }
            #pragma unroll
            for (int off = 1; off < 16; off <<= 1)
                rs += __shfl_xor(rs, off, 32);
            scale[r] = __expf(mi[r] - mnew);
            li[r] = li[r] * scale[r] + rs;
            mi[r] = mnew;
        }
        #pragma unroll
        for (int t = 0; t < 16; ++t)
            #pragma unroll
            for (int r = 0; r < 8; ++r)
                acc[t][r] = acc[t][r] * scale[r];

        // P (D-layout) -> per-wave LDS 16x64 row-major -> reload as A-frags.
        __bf16* pp = ldsP + (wave << 10);
        #pragma unroll
        for (int j = 0; j < 4; ++j)
            #pragma unroll
            for (int r = 0; r < 8; ++r) {
                const int M = r + (hh << 3);
                pp[M * 64 + (j << 4) + cc] = f2bf(p[j][r]);
            }
        asm volatile("s_wait_dscnt 0" ::: "memory");   // same-wave DS ordering
        v16bf aP[2];
        aP[0] = load_afrag(pp, 64, 0,  lane);
        aP[1] = load_afrag(pp, 64, 32, lane);

        // O += P(16x64) x V^T(64x256)
        #pragma unroll
        for (int kt = 0; kt < 2; ++kt)
            #pragma unroll
            for (int t = 0; t < 16; ++t) {
                v16bf bv16 = load_bfragT(ldsV, 72, t << 4, kt << 5, lane);
                acc[t] = __builtin_amdgcn_wmma_f32_16x16x32_bf16(
                    false, aP[kt], false, bv16, (short)0, acc[t], false, false);
            }
    }

    // Finalize: out[b][c][n] = x + gamma * O[n][c] / l[n]
    const float g = gamma[0];
    #pragma unroll
    for (int t = 0; t < 16; ++t) {
        #pragma unroll
        for (int r = 0; r < 8; ++r) {
            const int c = (t << 4) + cc;
            const int n = nq0 + r + (hh << 3);
            const size_t idx = ((size_t)(b << 8) + c) * 4096 + n;
            out[idx] = x[idx] + g * (acc[t][r] / li[r]);
        }
    }
}

// ---- Launch -----------------------------------------------------------------
extern "C" void kernel_launch(void* const* d_in, const int* in_sizes, int n_in,
                              void* d_out, int out_size, void* d_ws, size_t ws_size,
                              hipStream_t stream) {
    (void)in_sizes; (void)n_in; (void)out_size; (void)ws_size;
    const float* x     = (const float*)d_in[0];
    const float* wq    = (const float*)d_in[1];
    const float* bq    = (const float*)d_in[2];
    const float* wk    = (const float*)d_in[3];
    const float* bk    = (const float*)d_in[4];
    const float* wv    = (const float*)d_in[5];
    const float* bv    = (const float*)d_in[6];
    const float* gamma = (const float*)d_in[7];
    float* out = (float*)d_out;

    char* ws = (char*)d_ws;                                    // ~40.4 MiB
    __bf16* xb  = (__bf16*)(ws);                               // 16 MiB [b][n][c]
    __bf16* qws = (__bf16*)(ws + (size_t)16 * 1024 * 1024);    //  4 MiB [b][n][64]
    __bf16* kws = (__bf16*)(ws + (size_t)20 * 1024 * 1024);    //  4 MiB [b][n][64]
    __bf16* vws = (__bf16*)(ws + (size_t)24 * 1024 * 1024);    // 16 MiB [b][c][n]
    __bf16* wf  = (__bf16*)(ws + (size_t)40 * 1024 * 1024);    // 192 KiB fused W
    float*  bf  = (float*) (ws + (size_t)40 * 1024 * 1024 + 256 * 1024);

    k_prep<<<384, 256, 0, stream>>>(wq, bq, wk, bk, wv, bv, wf, bf);
    dim3 g1(64, 4, 8);
    k_cvt_t<<<g1, 256, 0, stream>>>(x, xb);
    dim3 g2(8, 24, 8);
    k_qkv<<<g2, 256, 0, stream>>>(xb, wf, bf, qws, kws, vws);
    dim3 g3(32, 8);
    k_attn<<<g3, 256, 0, stream>>>(x, qws, kws, vws, gamma, out);
}
